// Py_KAN_Fix_40724879900826
// MI455X (gfx1250) — compile-verified, matrix operand order checked by reading, not run on hardware
//
#include <hip/hip_runtime.h>
#include <hip/hip_bf16.h>

// ---------------------------------------------------------------------------
// KAN forward (2 layers + log_softmax) for MI455X / gfx1250, wave32 + WMMA.
//
// Layer math per edge:  y[b,o] = sum_i ( scale_base[o,i]*silu(x[b,i])
//                                      + scale_sp[o,i]*sum_p coef[o,i,p]*B_p(x[b,i]) ) + bias[o]
// Fused as f16 GEMM: per channel i, a 16-half feature vector
//   A: [silu, B0..B7, 0x7]   W: [scale_base, sp*coef0..7, 0x7]
// => 2 channels == one v_wmma_f32_16x16x32_f16 k-step.
// ---------------------------------------------------------------------------

typedef __attribute__((ext_vector_type(16))) _Float16 v16h;
typedef __attribute__((ext_vector_type(8)))  float    v8f;

#define B_ROWS   8192
#define IN1      784
#define HID      128
#define OUT2     10
#define P        8          // NUM + K = 5 + 3
#define CH       8          // channels per stage
#define KSTEPS   4          // wmma k-steps per stage (CH/2)
#define FRAGB    1056       // bytes per (kstep,mtile) A-frag block: 32 lanes*32B + 32B pad

// GEMM1 tiling
#define BM1      64
#define MT1      4          // m-tiles per block
#define NT1      8          // n-tiles = waves (covers N=128)
#define NSTAGE1  (IN1 / CH)     // 98
#define NKS1     (NSTAGE1 * KSTEPS) // 392

// GEMM2 tiling
#define BM2      128
#define MT2      8          // m-tiles = waves
#define NSTAGE2  (HID / CH)     // 16
#define NKS2     (NSTAGE2 * KSTEPS) // 64

// ---------------------------------------------------------------------------
// Cox-de Boor basis (order K=3, 12 knots) + SiLU. Reciprocal knot spans rd*
// are precomputed per block (grid-only, no per-element divides).
// ---------------------------------------------------------------------------
__device__ inline void kan_features(float x, const float* g,
                                    const float* rd1, const float* rd2,
                                    const float* rd3, float* e /*[9]*/) {
    float v0[11], v1[10], v2[9], v3[8];
#pragma unroll
    for (int j = 0; j < 11; ++j)
        v0[j] = ((x >= g[j]) && (x < g[j + 1])) ? 1.0f : 0.0f;
#pragma unroll
    for (int j = 0; j < 10; ++j)
        v1[j] = (x - g[j]) * rd1[j] * v0[j] + (g[j + 2] - x) * rd1[j + 1] * v0[j + 1];
#pragma unroll
    for (int j = 0; j < 9; ++j)
        v2[j] = (x - g[j]) * rd2[j] * v1[j] + (g[j + 3] - x) * rd2[j + 1] * v1[j + 1];
#pragma unroll
    for (int j = 0; j < 8; ++j)
        v3[j] = (x - g[j]) * rd3[j] * v2[j] + (g[j + 4] - x) * rd3[j + 1] * v2[j + 1];
    e[0] = x * __frcp_rn(1.0f + __expf(-x));   // silu
#pragma unroll
    for (int j = 0; j < 8; ++j) e[1 + j] = v3[j];
}

__device__ inline void fill_grid_lds(const float* __restrict__ grid,
                                     float* g, float* rd1, float* rd2, float* rd3) {
    int tid = threadIdx.x;
    if (tid < 12) g[tid] = grid[tid];
    __syncthreads();
    if (tid < 11)      rd1[tid] = 1.0f / (g[tid + 1] - g[tid]);
    else if (tid < 21) { int j = tid - 11; rd2[j] = 1.0f / (g[j + 2] - g[j]); }
    else if (tid < 30) { int j = tid - 21; rd3[j] = 1.0f / (g[j + 3] - g[j]); }
    __syncthreads();
}

// ---------------------------------------------------------------------------
// Weight packing into WMMA B-fragment order.
// Layout: wp[((kstep*NT + ntile)*32 + lane)*16 halves]  (32 B per lane, coalesced)
// B-frag (K x N = 32 x 16), mirroring the A layout:
//   lane l<16 : N=l,     halves 0..7 = K0..7  (even ch elems 0..7), 8..15 = K16..23 (odd ch 0..7)
//   lane l>=16: N=l-16,  halves 0..7 = K8..15 (even ch elems 8..15), 8..15 = K24..31 (odd ch 8..15)
// channel element vector: [scale_base, sp*coef0..7, 0,0,0,0,0,0,0]
// ---------------------------------------------------------------------------
__global__ void kan_pack_w(const float* __restrict__ sb, const float* __restrict__ sp,
                           const float* __restrict__ coef, _Float16* __restrict__ wp,
                           int in_dim, int n_tiles, int n_real, int total) {
    int idx = blockIdx.x * 256 + threadIdx.x;
    if (idx >= total) return;
    int lane = idx & 31;
    int nt   = (idx >> 5) % n_tiles;
    int sg   = idx / (32 * n_tiles);          // global k-step
    int n    = nt * 16 + (lane & 15);
    int hi   = (lane >> 4) * 8;               // 0 -> elems 0..7, 1 -> elems 8..15

    _Float16 hv[16] __attribute__((aligned(16)));
#pragma unroll
    for (int half = 0; half < 2; ++half) {    // even / odd channel of this k-step
        int c = sg * 2 + half;
        float e[16];
#pragma unroll
        for (int q = 0; q < 16; ++q) e[q] = 0.0f;
        if (n < n_real) {
            float b = sb[n * in_dim + c];
            float s = sp[n * in_dim + c];
            e[0] = b;
#pragma unroll
            for (int p = 0; p < P; ++p) e[1 + p] = s * coef[(n * in_dim + c) * P + p];
        }
#pragma unroll
        for (int q = 0; q < 8; ++q) hv[half * 8 + q] = (_Float16)e[hi + q];
    }
    *(int4*)&wp[(size_t)idx * 16]     = *(int4*)&hv[0];
    *(int4*)&wp[(size_t)idx * 16 + 8] = *(int4*)&hv[8];
}

// ---------------------------------------------------------------------------
// GEMM1: h[8192,128] = A1(x) * W1^T + bias1, basis computed on the fly.
// 128 blocks x 256 threads (8 waves). Wave w owns n-tile w (16 cols) and all
// MT1=4 m-tiles (64 rows) -> zero redundant B-fragment traffic.
// ---------------------------------------------------------------------------
__global__ __launch_bounds__(256) void kan_gemm1(const float* __restrict__ x,
                                                 const float* __restrict__ grid,
                                                 const _Float16* __restrict__ w1p,
                                                 const float* __restrict__ bias1,
                                                 float* __restrict__ h) {
    __shared__ float g[12], rd1[11], rd2[10], rd3[9];
    __shared__ __align__(32) char afrag[KSTEPS * MT1 * FRAGB];

    fill_grid_lds(grid, g, rd1, rd2, rd3);

    const int tid  = threadIdx.x;
    const int wave = tid >> 5;
    const int lane = tid & 31;
    const int row0 = blockIdx.x * BM1;

    v8f acc[MT1] = {};

    for (int stage = 0; stage < NSTAGE1; ++stage) {
        const int c0 = stage * CH;
        __syncthreads();                       // previous consume done
        // -------- produce: spline features -> LDS in A-fragment layout ------
#pragma unroll
        for (int it = 0; it < (BM1 * CH) / 256; ++it) {
            int t  = it * 256 + tid;
            int m  = t >> 3;                   // 0..63
            int il = t & 7;                    // local channel
            float xv = x[(size_t)(row0 + m) * IN1 + c0 + il];
            float e[9];
            kan_features(xv, g, rd1, rd2, rd3, e);
            _Float16 hv[16] __attribute__((aligned(16)));
#pragma unroll
            for (int q = 0; q < 9; ++q) hv[q] = (_Float16)e[q];
#pragma unroll
            for (int q = 9; q < 16; ++q) hv[q] = (_Float16)0.0f;
            int ks = il >> 1, odd = il & 1, mt = m >> 4, ml = m & 15;
            char* base = &afrag[(ks * MT1 + mt) * FRAGB + odd * 16];
            *(int4*)(base + ml * 32)        = *(int4*)&hv[0];  // K low half
            *(int4*)(base + (ml + 16) * 32) = *(int4*)&hv[8];  // K high half
        }
        __syncthreads();
        // -------- consume: WMMA ---------------------------------------------
        // Prefetch next stage's B fragments for this wave (global_prefetch_b8)
        if (stage + 1 < NSTAGE1)
            __builtin_prefetch(&w1p[((size_t)((stage + 1) * KSTEPS * NT1 + wave) * 32 + lane) * 16], 0, 3);
#pragma unroll
        for (int ks = 0; ks < KSTEPS; ++ks) {
            int sg = stage * KSTEPS + ks;
            v16h b = *(const v16h*)&w1p[((size_t)(sg * NT1 + wave) * 32 + lane) * 16];
#pragma unroll
            for (int mt = 0; mt < MT1; ++mt) {
                v16h a = *(const v16h*)&afrag[(ks * MT1 + mt) * FRAGB + lane * 32];
                acc[mt] = __builtin_amdgcn_wmma_f32_16x16x32_f16(
                    false, a, false, b, (short)0, acc[mt], false, false);
            }
        }
    }

    // -------- epilogue: C layout -> h (f32) + bias ---------------------------
    const int n    = wave * 16 + (lane & 15);
    const float bv = bias1[n];
    const int mro  = (lane >> 4) * 8;
#pragma unroll
    for (int mt = 0; mt < MT1; ++mt)
#pragma unroll
        for (int r = 0; r < 8; ++r)
            h[(size_t)(row0 + mt * 16 + mro + r) * HID + n] = acc[mt][r] + bv;
}

// ---------------------------------------------------------------------------
// GEMM2: o[8192,16] = A2(h) * W2^T + bias2 (cols 10..15 are zero-padded).
// 64 blocks x 256 threads; wave w owns m-tile w, single n-tile.
// ---------------------------------------------------------------------------
__global__ __launch_bounds__(256) void kan_gemm2(const float* __restrict__ hin,
                                                 const float* __restrict__ grid,
                                                 const _Float16* __restrict__ w2p,
                                                 const float* __restrict__ bias2,
                                                 float* __restrict__ o) {
    __shared__ float g[12], rd1[11], rd2[10], rd3[9];
    __shared__ __align__(32) char afrag[KSTEPS * MT2 * FRAGB];

    fill_grid_lds(grid, g, rd1, rd2, rd3);

    const int tid  = threadIdx.x;
    const int wave = tid >> 5;
    const int lane = tid & 31;
    const int row0 = blockIdx.x * BM2;

    v8f acc = {};

    for (int stage = 0; stage < NSTAGE2; ++stage) {
        const int c0 = stage * CH;
        __syncthreads();
#pragma unroll
        for (int it = 0; it < (BM2 * CH) / 256; ++it) {
            int t  = it * 256 + tid;
            int m  = t >> 3;                   // 0..127
            int il = t & 7;
            float xv = hin[(size_t)(row0 + m) * HID + c0 + il];
            float e[9];
            kan_features(xv, g, rd1, rd2, rd3, e);
            _Float16 hv[16] __attribute__((aligned(16)));
#pragma unroll
            for (int q = 0; q < 9; ++q) hv[q] = (_Float16)e[q];
#pragma unroll
            for (int q = 9; q < 16; ++q) hv[q] = (_Float16)0.0f;
            int ks = il >> 1, odd = il & 1, mt = m >> 4, ml = m & 15;
            char* base = &afrag[(ks * MT2 + mt) * FRAGB + odd * 16];
            *(int4*)(base + ml * 32)        = *(int4*)&hv[0];
            *(int4*)(base + (ml + 16) * 32) = *(int4*)&hv[8];
        }
        __syncthreads();
#pragma unroll
        for (int ks = 0; ks < KSTEPS; ++ks) {
            int sg = stage * KSTEPS + ks;
            v16h b = *(const v16h*)&w2p[((size_t)sg * 32 + lane) * 16];
            v16h a = *(const v16h*)&afrag[(ks * MT2 + wave) * FRAGB + lane * 32];
            acc = __builtin_amdgcn_wmma_f32_16x16x32_f16(
                false, a, false, b, (short)0, acc, false, false);
        }
    }

    const int n   = lane & 15;
    const int mro = (lane >> 4) * 8;
    const float bv = (n < OUT2) ? bias2[n] : 0.0f;
#pragma unroll
    for (int r = 0; r < 8; ++r)
        o[(size_t)(row0 + wave * 16 + mro + r) * 16 + n] = acc[r] + bv;
}

// ---------------------------------------------------------------------------
// Row-wise log_softmax over the 10 valid columns.
// ---------------------------------------------------------------------------
__global__ void kan_logsoftmax(const float* __restrict__ o, float* __restrict__ out) {
    int r = blockIdx.x * blockDim.x + threadIdx.x;
    if (r >= B_ROWS) return;
    float v[OUT2];
    float m = -3.4e38f;
#pragma unroll
    for (int j = 0; j < OUT2; ++j) { v[j] = o[(size_t)r * 16 + j]; m = fmaxf(m, v[j]); }
    float s = 0.0f;
#pragma unroll
    for (int j = 0; j < OUT2; ++j) s += __expf(v[j] - m);
    float ls = __logf(s);
#pragma unroll
    for (int j = 0; j < OUT2; ++j) out[(size_t)r * OUT2 + j] = v[j] - m - ls;
}

// ---------------------------------------------------------------------------
// Launch. Inputs (setup_inputs order):
// 0:x 1:grid 2:scale_base1 3:scale_sp1 4:coef1 5:bias1
// 6:scale_base2 7:scale_sp2 8:coef2 9:bias2
// ---------------------------------------------------------------------------
extern "C" void kernel_launch(void* const* d_in, const int* in_sizes, int n_in,
                              void* d_out, int out_size, void* d_ws, size_t ws_size,
                              hipStream_t stream) {
    const float* x     = (const float*)d_in[0];
    const float* grid  = (const float*)d_in[1];
    const float* sb1   = (const float*)d_in[2];
    const float* sp1   = (const float*)d_in[3];
    const float* coef1 = (const float*)d_in[4];
    const float* bias1 = (const float*)d_in[5];
    const float* sb2   = (const float*)d_in[6];
    const float* sp2   = (const float*)d_in[7];
    const float* coef2 = (const float*)d_in[8];
    const float* bias2 = (const float*)d_in[9];

    char* ws = (char*)d_ws;
    const size_t W1P_BYTES = (size_t)NKS1 * NT1 * 32 * 32;   // 3,211,264
    const size_t W2P_BYTES = (size_t)NKS2 * 1   * 32 * 32;   //    65,536
    const size_t H_BYTES   = (size_t)B_ROWS * HID * 4;       // 4,194,304
    _Float16* w1p = (_Float16*)ws;
    _Float16* w2p = (_Float16*)(ws + W1P_BYTES);
    float*    h   = (float*)(ws + W1P_BYTES + W2P_BYTES);
    float*    o   = (float*)(ws + W1P_BYTES + W2P_BYTES + H_BYTES);
    float*    out = (float*)d_out;

    int tot1 = NKS1 * NT1 * 32;   // 100352
    int tot2 = NKS2 * 1   * 32;   // 2048
    kan_pack_w<<<(tot1 + 255) / 256, 256, 0, stream>>>(sb1, sp1, coef1, w1p, IN1, NT1, HID, tot1);
    kan_pack_w<<<(tot2 + 255) / 256, 256, 0, stream>>>(sb2, sp2, coef2, w2p, HID, 1, OUT2, tot2);
    kan_gemm1<<<B_ROWS / BM1, 256, 0, stream>>>(x, grid, w1p, bias1, h);
    kan_gemm2<<<B_ROWS / BM2, 256, 0, stream>>>(h, grid, w2p, bias2, o);
    kan_logsoftmax<<<B_ROWS / 256, 256, 0, stream>>>(o, out);
}
